// SourceAwareContrastiveLoss_71579924955708
// MI455X (gfx1250) — matrix operand order repaired and checked by reading, not compiled, not run on hardware
//
#include <hip/hip_runtime.h>

#define N_ROWS   8192
#define DIM      512
#define M_BLOCK  64
#define N_TILE   64
#define K_CHUNK  32
#define THREADS  128
#define CSLICES  4
#define SLICE_COLS (N_ROWS / CSLICES)
#define LDA      520                            // padded bf16 stride (bank-conflict free)
#define TILE_BYTES (M_BLOCK * LDA * 2)          // 66,560 B per hi/lo plane
#define SMEM_BYTES (4 * TILE_BYTES)             // 266,240 B  (<= 320 KB WGP LDS)
#define PLANE_BYTES ((size_t)N_ROWS * LDA * 2)  // one full bf16 plane: 8,519,680 B
#define TILE_V16   (TILE_BYTES / 16)            // 4160 x 16B lines per plane

typedef __attribute__((ext_vector_type(16))) __bf16 v16bf;
typedef __attribute__((ext_vector_type(8)))  float  v8f;
typedef __attribute__((ext_vector_type(4)))  int    v4i;
typedef __attribute__((address_space(1))) v4i g_v4i;   // global (HIP __device__) space
typedef __attribute__((address_space(3))) v4i s_v4i;   // LDS space

union Frag { v16bf v; unsigned u[8]; uint4 q[2]; };
static_assert(sizeof(v16bf) == 32, "v16bf must be 8 dwords");
static_assert(TILE_BYTES % 16 == 0 && (2 * TILE_V16) % THREADS == 0, "copy tiling");

__device__ __forceinline__ unsigned short f2bf(float f) {       // RNE fp32 -> bf16
  unsigned u = __float_as_uint(f);
  u += 0x7fffu + ((u >> 16) & 1u);
  return (unsigned short)(u >> 16);
}
__device__ __forceinline__ float bf2f(unsigned short s) {
  return __uint_as_float(((unsigned)s) << 16);
}

#if defined(__has_builtin)
#if __has_builtin(__builtin_amdgcn_global_load_async_to_lds_b128)
#define SACL_ASYNC_CP 1
#endif
#endif

// 16-byte global->LDS line copy (async DMA when available).
__device__ __forceinline__ void cp16(char* ldsDst, const char* gSrc) {
#ifdef SACL_ASYNC_CP
  __builtin_amdgcn_global_load_async_to_lds_b128(
      (g_v4i*)gSrc, (s_v4i*)ldsDst, 0, 0);
#else
  *reinterpret_cast<uint4*>(ldsDst) = *reinterpret_cast<const uint4*>(gSrc);
#endif
}
__device__ __forceinline__ void cp_wait() {
#ifdef SACL_ASYNC_CP
  asm volatile("s_wait_asynccnt 0x0" ::: "memory");
#endif
}

// One-time pre-pass: split both fp32 matrices into padded bf16 hi/lo planes.
__global__ void __launch_bounds__(256)
sacl_convert_kernel(const float* __restrict__ img, const float* __restrict__ txt,
                    char* __restrict__ ws) {
  unsigned* iHi = (unsigned*)(ws);
  unsigned* iLo = (unsigned*)(ws + PLANE_BYTES);
  unsigned* tHi = (unsigned*)(ws + 2 * PLANE_BYTES);
  unsigned* tLo = (unsigned*)(ws + 3 * PLANE_BYTES);
  const int total = N_ROWS * (DIM / 4);
  for (int idx = blockIdx.x * blockDim.x + threadIdx.x; idx < total;
       idx += gridDim.x * blockDim.x) {
    const int row = idx >> 7;                 // 128 float4 per row
    const int kq  = (idx & 127) << 2;
    const size_t gsrc = (size_t)row * DIM + kq;
    const size_t dsto = ((size_t)row * LDA + kq) >> 1;   // uint index (2 bf16/uint)
#pragma unroll
    for (int m = 0; m < 2; ++m) {
      const float4 f = *reinterpret_cast<const float4*>((m ? txt : img) + gsrc);
      const unsigned short h0 = f2bf(f.x), h1 = f2bf(f.y), h2 = f2bf(f.z), h3 = f2bf(f.w);
      const unsigned short l0 = f2bf(f.x - bf2f(h0)), l1 = f2bf(f.y - bf2f(h1));
      const unsigned short l2 = f2bf(f.z - bf2f(h2)), l3 = f2bf(f.w - bf2f(h3));
      unsigned* hi = m ? tHi : iHi;
      unsigned* lo = m ? tLo : iLo;
      hi[dsto]     = (unsigned)h0 | ((unsigned)h1 << 16);
      hi[dsto + 1] = (unsigned)h2 | ((unsigned)h3 << 16);
      lo[dsto]     = (unsigned)l0 | ((unsigned)l1 << 16);
      lo[dsto + 1] = (unsigned)l2 | ((unsigned)l3 << 16);
    }
  }
}

__global__ void __launch_bounds__(THREADS, 1)
sacl_main_kernel(const char* __restrict__ ws, const float* __restrict__ scalep,
                 const long long* __restrict__ lab, float4* __restrict__ part) {
  extern __shared__ char smem[];
  char* aHi = smem;                           // + TILE_BYTES = aLo
  char* bHi = smem + 2 * (size_t)TILE_BYTES;  // + TILE_BYTES = bLo

  const int tid     = threadIdx.x;
  const int lane    = tid & 31;
  const int wave    = tid >> 5;
  const int nl      = lane & 15;   // column-in-tile / A-row-in-frag
  const int lh      = lane >> 4;   // lane half (K split per ISA layout)
  const int rowBase = blockIdx.x * M_BLOCK;
  const int jStart  = blockIdx.y * SLICE_COLS;
  const float scale = scalep[0];

  const char* gImg = ws + (size_t)rowBase * LDA * 2;       // imgHi plane (+PLANE_BYTES = lo)
  const char* gTxt = ws + 2 * PLANE_BYTES;                 // txtHi plane (+PLANE_BYTES = lo)

  // Stage this block's 64 image rows (hi+lo planes) — pure contiguous copy.
  for (int idx = tid; idx < 2 * TILE_V16; idx += THREADS) {
    const int p = (idx >= TILE_V16);
    const size_t o = (size_t)(idx - p * TILE_V16) * 16;
    cp16(aHi + (size_t)p * TILE_BYTES + o, gImg + (size_t)p * PLANE_BYTES + o);
  }
  cp_wait();

  long long rowLab[8];
#pragma unroll
  for (int r = 0; r < 8; ++r)
    rowLab[r] = lab[rowBase + wave * 16 + (lh << 3) + r];

  float runMax[8], runSum[8], mSum[8], cnt[8];
#pragma unroll
  for (int r = 0; r < 8; ++r) { runMax[r] = -3.0e38f; runSum[r] = 0.f; mSum[r] = 0.f; cnt[r] = 0.f; }

  // Loop-invariant fragment base pointers: all inner-loop DS offsets become immediates.
  const char* aHiB = aHi + (((size_t)(wave * 16 + nl) * LDA + (lh << 3)) << 1);
  const char* aLoB = aHiB + TILE_BYTES;
  const char* bB[4];
#pragma unroll
  for (int t = 0; t < 4; ++t)
    bB[t] = bHi + (((size_t)(t * 16 + nl) * LDA + (lh << 3)) << 1);

  for (int jt = jStart; jt < jStart + SLICE_COLS; jt += N_TILE) {
    __syncthreads();                       // previous tile fully consumed
    const char* gB = gTxt + (size_t)jt * LDA * 2;
    for (int idx = tid; idx < 2 * TILE_V16; idx += THREADS) {
      const int p = (idx >= TILE_V16);
      const size_t o = (size_t)(idx - p * TILE_V16) * 16;
      cp16(bHi + (size_t)p * TILE_BYTES + o, gB + (size_t)p * PLANE_BYTES + o);
    }
    cp_wait();
    __syncthreads();                       // staged tile visible to all waves

    v8f acc[4];
#pragma unroll
    for (int t = 0; t < 4; ++t) acc[t] = (v8f){0.f,0.f,0.f,0.f,0.f,0.f,0.f,0.f};

#pragma unroll 2
    for (int kc = 0; kc < DIM / K_CHUNK; ++kc) {
      const int kb = kc << 6;              // kc * 64 bytes
      Frag ah, al, bh[4], bl[4];
      ah.q[0] = *(const uint4*)(aHiB + kb);      ah.q[1] = *(const uint4*)(aHiB + kb + 32);
      al.q[0] = *(const uint4*)(aLoB + kb);      al.q[1] = *(const uint4*)(aLoB + kb + 32);
#pragma unroll
      for (int t = 0; t < 4; ++t) {
        bh[t].q[0] = *(const uint4*)(bB[t] + kb);
        bh[t].q[1] = *(const uint4*)(bB[t] + kb + 32);
        bl[t].q[0] = *(const uint4*)(bB[t] + TILE_BYTES + kb);
        bl[t].q[1] = *(const uint4*)(bB[t] + TILE_BYTES + kb + 32);
      }
      // Three passes over t: consecutive WMMAs never share an accumulator (hide XDL RAW).
#pragma unroll
      for (int t = 0; t < 4; ++t)
        acc[t] = __builtin_amdgcn_wmma_f32_16x16x32_bf16(false, ah.v, false, bh[t].v, (short)0, acc[t], false, false);
#pragma unroll
      for (int t = 0; t < 4; ++t)
        acc[t] = __builtin_amdgcn_wmma_f32_16x16x32_bf16(false, ah.v, false, bl[t].v, (short)0, acc[t], false, false);
#pragma unroll
      for (int t = 0; t < 4; ++t)
        acc[t] = __builtin_amdgcn_wmma_f32_16x16x32_bf16(false, al.v, false, bh[t].v, (short)0, acc[t], false, false);
    }

    // Online softmax + masked-sum epilogue (per lane column slot).
    long long lj[4];
#pragma unroll
    for (int t = 0; t < 4; ++t) lj[t] = lab[jt + t * 16 + nl];
#pragma unroll
    for (int r = 0; r < 8; ++r) {
      float v[4];
#pragma unroll
      for (int t = 0; t < 4; ++t) v[t] = scale * acc[t][r];
      const float lm = fmaxf(fmaxf(v[0], v[1]), fmaxf(v[2], v[3]));
      const float le = __expf(v[0] - lm) + __expf(v[1] - lm) + __expf(v[2] - lm) + __expf(v[3] - lm);
      const float nm = fmaxf(runMax[r], lm);
      runSum[r] = runSum[r] * __expf(runMax[r] - nm) + le * __expf(lm - nm);
      runMax[r] = nm;
      const int i = rowBase + wave * 16 + (lh << 3) + r;
#pragma unroll
      for (int t = 0; t < 4; ++t)
        if ((lj[t] == rowLab[r]) && (i != jt + t * 16 + nl)) { mSum[r] += v[t]; cnt[r] += 1.f; }
    }
  }

  // Reduce across the 16 lanes of each half (column direction of the C tile).
#pragma unroll
  for (int r = 0; r < 8; ++r) {
    for (int off = 1; off < 16; off <<= 1) {
      const float om = __shfl_xor(runMax[r], off, 32);
      const float os = __shfl_xor(runSum[r], off, 32);
      const float nm = fmaxf(runMax[r], om);
      runSum[r] = runSum[r] * __expf(runMax[r] - nm) + os * __expf(om - nm);
      runMax[r] = nm;
      mSum[r] += __shfl_xor(mSum[r], off, 32);
      cnt[r]  += __shfl_xor(cnt[r], off, 32);
    }
  }
  if (nl == 0) {
#pragma unroll
    for (int r = 0; r < 8; ++r) {
      const int i = rowBase + wave * 16 + (lh << 3) + r;
      float4 p; p.x = runMax[r]; p.y = runSum[r]; p.z = mSum[r]; p.w = cnt[r];
      part[(size_t)i * CSLICES + blockIdx.y] = p;
    }
  }
}

__global__ void sacl_zero_kernel(float* out) { out[0] = 0.f; }

__global__ void __launch_bounds__(256)
sacl_finalize_kernel(const float4* __restrict__ part, float* __restrict__ out) {
  __shared__ float blockAcc;
  if (threadIdx.x == 0) blockAcc = 0.f;
  __syncthreads();
  const int i = blockIdx.x * blockDim.x + threadIdx.x;
  float m = -3.0e38f, s = 0.f, ms = 0.f, c = 0.f;
#pragma unroll
  for (int k = 0; k < CSLICES; ++k) {
    const float4 p = part[(size_t)i * CSLICES + k];
    const float nm = fmaxf(m, p.x);
    s = s * __expf(m - nm) + p.y * __expf(p.x - nm);
    m = nm;
    ms += p.z; c += p.w;
  }
  float contrib = 0.f;
  if (c > 0.f) contrib = (ms - c * (m + __logf(s))) / c;
  for (int off = 16; off; off >>= 1) contrib += __shfl_xor(contrib, off, 32);
  if ((threadIdx.x & 31) == 0) atomicAdd(&blockAcc, contrib);
  __syncthreads();
  if (threadIdx.x == 0) atomicAdd(out, -blockAcc / (float)N_ROWS);
}

extern "C" void kernel_launch(void* const* d_in, const int* in_sizes, int n_in,
                              void* d_out, int out_size, void* d_ws, size_t ws_size,
                              hipStream_t stream) {
  (void)in_sizes; (void)n_in; (void)out_size; (void)ws_size;
  const float*     img = (const float*)d_in[0];
  const float*     txt = (const float*)d_in[1];
  const float*     scl = (const float*)d_in[2];
  const long long* lab = (const long long*)d_in[3];  // jnp.int64 labels
  float* out = (float*)d_out;
  char*  ws  = (char*)d_ws;                           // 4 bf16 planes (~34 MB) + partials
  float4* part = (float4*)(ws + 4 * PLANE_BYTES);     // 8192*4*16 B = 512 KB

  (void)hipFuncSetAttribute((const void*)sacl_main_kernel,
                            hipFuncAttributeMaxDynamicSharedMemorySize, SMEM_BYTES);

  sacl_zero_kernel<<<1, 1, 0, stream>>>(out);
  sacl_convert_kernel<<<1024, 256, 0, stream>>>(img, txt, ws);
  dim3 grid(N_ROWS / M_BLOCK, CSLICES);               // 128 x 4 workgroups
  sacl_main_kernel<<<grid, THREADS, SMEM_BYTES, stream>>>(ws, scl, lab, part);
  sacl_finalize_kernel<<<N_ROWS / 256, 256, 0, stream>>>(part, out);
}
